// LSTMTaggerGen102_65481071397180
// MI455X (gfx1250) — matrix operand order repaired and checked by reading, not compiled, not run on hardware
//
#include <hip/hip_runtime.h>
#include <cstdint>
#include <cmath>

#define NQ     4
#define SLEN   256
#define BATCH  512
#define EDIM   256
#define DDIM   260
#define TTAGS  64

typedef __attribute__((ext_vector_type(16))) __bf16 v16bf;
typedef __attribute__((ext_vector_type(8)))  float  v8f;

union BfPack { v16bf v; unsigned short u[16]; };

__device__ __forceinline__ unsigned short f32_to_bf16(float x) {
  unsigned int b = __float_as_uint(x);
  b += 0x7fffu + ((b >> 16) & 1u);           // round-to-nearest-even
  return (unsigned short)(b >> 16);
}
__device__ __forceinline__ float bf16_to_f32(unsigned short h) {
  return __uint_as_float(((unsigned int)h) << 16);
}

// ---------------------------------------------------------------------------
// Kernel 1: pre[s,b,n] = sum_e lin_w[n>>2, n&3, e] * emb[sentence[s,b], e] + lin_b[n]
// (x-part of the LSTM gate pre-activations, hoisted out of the scan)
// Split-bf16 WMMA: x = hi + lo, acc += Ahi*Bhi + Ahi*Blo + Alo*Bhi  (f32 acc)
// One wave per 16-row tile of the (131072 x 256) * (256 x 16) GEMM.
// ---------------------------------------------------------------------------
__global__ void __launch_bounds__(256)
k_pregemm(const int* __restrict__ sentence, const float* __restrict__ emb,
          const float* __restrict__ lin_w, const float* __restrict__ lin_b,
          float* __restrict__ pre)
{
  const int lane = threadIdx.x & 31;
  const int wave = threadIdx.x >> 5;
  const int tile = blockIdx.x * 8 + wave;       // 8192 tiles total
  const int r0   = tile * 16;                   // first flattened (s,b) row
  const int m    = lane & 15;                   // A row owned by this lane
  const int half = lane >> 4;
  const int n    = lane & 15;                   // output column (g*4+q)

  const int tok = sentence[r0 + m];
  const float* arow = emb + (size_t)tok * EDIM;
  const float* wcol = lin_w + (n >> 2) * (NQ * DDIM) + (n & 3) * DDIM;

  v8f acc = {};
  for (int kc = 0; kc < 8; ++kc) {
    // A (16x32 bf16) layout: half 0 -> K {0..7, 16..23}; half 1 -> K {8..15, 24..31}
    const int ka0 = kc * 32 + half * 8;
    const int ka1 = ka0 + 16;
    // B (32x16 bf16) layout: half 0 -> K {0..7, 16..23}; half 1 -> K {8..15, 24..31}
    const int kb  = kc * 32 + half * 8;

    float ax[16], bx[16];
    *(float4*)&ax[0]  = *(const float4*)(arow + ka0);
    *(float4*)&ax[4]  = *(const float4*)(arow + ka0 + 4);
    *(float4*)&ax[8]  = *(const float4*)(arow + ka1);
    *(float4*)&ax[12] = *(const float4*)(arow + ka1 + 4);
    *(float4*)&bx[0]  = *(const float4*)(wcol + kb);
    *(float4*)&bx[4]  = *(const float4*)(wcol + kb + 4);
    *(float4*)&bx[8]  = *(const float4*)(wcol + kb + 16);
    *(float4*)&bx[12] = *(const float4*)(wcol + kb + 20);

    BfPack ahi, alo, bhi, blo;
#pragma unroll
    for (int e = 0; e < 16; ++e) {
      unsigned short ha = f32_to_bf16(ax[e]);
      ahi.u[e] = ha;
      alo.u[e] = f32_to_bf16(ax[e] - bf16_to_f32(ha));
      unsigned short hb = f32_to_bf16(bx[e]);
      bhi.u[e] = hb;
      blo.u[e] = f32_to_bf16(bx[e] - bf16_to_f32(hb));
    }
    acc = __builtin_amdgcn_wmma_f32_16x16x32_bf16(false, ahi.v, false, bhi.v,
                                                  (short)0, acc, false, false);
    acc = __builtin_amdgcn_wmma_f32_16x16x32_bf16(false, ahi.v, false, blo.v,
                                                  (short)0, acc, false, false);
    acc = __builtin_amdgcn_wmma_f32_16x16x32_bf16(false, alo.v, false, bhi.v,
                                                  (short)0, acc, false, false);
  }

  const float bias = lin_b[n];
#pragma unroll
  for (int r = 0; r < 8; ++r) {
    const int Mrow = r + half * 8;             // C/D layout: VGPR r -> row r (+8)
    pre[(size_t)(r0 + Mrow) * 16 + n] = acc[r] + bias;
  }
}

// ---------------------------------------------------------------------------
// Kernel 2: build M[g][w] = Re(U† D_w U) (16x16 real symmetric) per gate.
// U = trainable ring ∘ fixed random layer. One block per gate g; lanes 0..15
// each evolve one column of U in registers, then reduce via LDS.
// ---------------------------------------------------------------------------
struct QOp  { int kind; int w0; int w1; float theta; };
struct QOps { QOp op[4][20]; };

__device__ __forceinline__ void q_rx(float* re, float* im, int w, float t) {
  float c = cosf(0.5f * t), s = sinf(0.5f * t);
  int msk = 8 >> w;
  for (int i = 0; i < 16; ++i) if (!(i & msk)) {
    int i1 = i | msk;
    float r0 = re[i], m0 = im[i], r1 = re[i1], m1 = im[i1];
    re[i]  =  c * r0 + s * m1;  im[i]  =  c * m0 - s * r1;
    re[i1] =  s * m0 + c * r1;  im[i1] = -s * r0 + c * m1;
  }
}
__device__ __forceinline__ void q_ry(float* re, float* im, int w, float t) {
  float c = cosf(0.5f * t), s = sinf(0.5f * t);
  int msk = 8 >> w;
  for (int i = 0; i < 16; ++i) if (!(i & msk)) {
    int i1 = i | msk;
    float r0 = re[i], m0 = im[i], r1 = re[i1], m1 = im[i1];
    re[i]  = c * r0 - s * r1;  im[i]  = c * m0 - s * m1;
    re[i1] = s * r0 + c * r1;  im[i1] = s * m0 + c * m1;
  }
}
__device__ __forceinline__ void q_rz(float* re, float* im, int w, float t) {
  float c = cosf(0.5f * t), s = sinf(0.5f * t);
  int msk = 8 >> w;
  for (int i = 0; i < 16; ++i) {
    float r = re[i], m = im[i];
    if (!(i & msk)) { re[i] = c * r + s * m;  im[i] = c * m - s * r; }   // * e^{-it/2}
    else            { re[i] = c * r - s * m;  im[i] = c * m + s * r; }   // * e^{+it/2}
  }
}
__device__ __forceinline__ void q_cnot(float* re, float* im, int w0, int w1) {
  int mc = 8 >> w0, mt = 8 >> w1;
  for (int i = 0; i < 16; ++i) if ((i & mc) && !(i & mt)) {
    int i1 = i | mt;
    float r = re[i], m = im[i];
    re[i] = re[i1]; im[i] = im[i1]; re[i1] = r; im[i1] = m;
  }
}
__device__ __forceinline__ void q_crx(float* re, float* im, int w0, int w1, float t) {
  float c = cosf(0.5f * t), s = sinf(0.5f * t);
  int mc = 8 >> w0, mt = 8 >> w1;
  for (int i = 0; i < 16; ++i) if ((i & mc) && !(i & mt)) {
    int i1 = i | mt;
    float r0 = re[i], m0 = im[i], r1 = re[i1], m1 = im[i1];
    re[i]  =  c * r0 + s * m1;  im[i]  =  c * m0 - s * r1;
    re[i1] =  s * m0 + c * r1;  im[i1] = -s * r0 + c * m1;
  }
}

__global__ void k_build_m(QOps ops, const float* __restrict__ qrx,
                          const float* __restrict__ qry, const float* __restrict__ qrz,
                          const float* __restrict__ qcrx, float* __restrict__ Mout)
{
  __shared__ float Ure[16][16], Uim[16][16];
  const int g = blockIdx.x;
  const int j = threadIdx.x;
  if (j < 16) {
    float re[16], im[16];
    for (int i = 0; i < 16; ++i) { re[i] = (i == j) ? 1.f : 0.f; im[i] = 0.f; }
    for (int o = 0; o < 20; ++o) {
      QOp p = ops.op[g][o];
      if      (p.kind == 0) q_rx(re, im, p.w0, p.theta);
      else if (p.kind == 1) q_ry(re, im, p.w0, p.theta);
      else if (p.kind == 2) q_rz(re, im, p.w0, p.theta);
      else                  q_cnot(re, im, p.w0, p.w1);
    }
    for (int w = 0; w < 4; ++w) {
      q_rx(re, im, w, qrx[g * 4 + w]);
      q_ry(re, im, w, qry[g * 4 + w]);
      q_rz(re, im, w, qrz[g * 4 + w]);
      q_crx(re, im, w, (w + 1) & 3, qcrx[g * 4 + w]);
    }
    for (int i = 0; i < 16; ++i) { Ure[i][j] = re[i]; Uim[i][j] = im[i]; }
  }
  __syncthreads();
  if (j < 16) {
    for (int w = 0; w < 4; ++w)
      for (int k = 0; k < 16; ++k) {
        float acc = 0.f;
        for (int i = 0; i < 16; ++i) {
          float d = ((i >> (3 - w)) & 1) ? -1.f : 1.f;
          acc += d * (Ure[i][j] * Ure[i][k] + Uim[i][j] * Uim[i][k]);
        }
        Mout[((g * 4 + w) * 16 + j) * 16 + k] = acc;
      }
  }
}

// ---------------------------------------------------------------------------
// Kernel 3: the sequential LSTM scan. Independent per batch element; 4 lanes
// per element (one per gate f/i/g/o), exchange via shfl_xor butterflies.
// M matrices live in LDS with a 16-float per-gate pad (kills 4-way bank
// conflicts since 1024 % 64 == 0).
// ---------------------------------------------------------------------------
__device__ __forceinline__ float pick4(float own, float v1, float v2, float v3, int sel) {
  return sel == 0 ? own : (sel == 1 ? v1 : (sel == 2 ? v2 : v3));
}

__global__ void __launch_bounds__(256)
k_scan(const float* __restrict__ pre, const float* __restrict__ lin_w,
       const float* __restrict__ Mglob, float* __restrict__ outsH)
{
  __shared__ float Ms[4 * 1040];
  __shared__ float Wh[64];
  for (int idx = threadIdx.x; idx < 4096; idx += blockDim.x)
    Ms[(idx >> 10) * 1040 + (idx & 1023)] = Mglob[idx];
  if (threadIdx.x < 64) {
    int nn = threadIdx.x >> 2, jj = threadIdx.x & 3;
    Wh[threadIdx.x] = lin_w[(nn >> 2) * (NQ * DDIM) + (nn & 3) * DDIM + EDIM + jj];
  }
  __syncthreads();

  const int t = blockIdx.x * blockDim.x + threadIdx.x;   // 0..2047
  const int b = t >> 2;                                  // batch element
  const int g = t & 3;                                   // gate: 0=f 1=i 2=g 3=o
  const float* Mg  = Ms + g * 1040;
  const float* WhG = Wh + g * 16;

  float h[4] = {0.f, 0.f, 0.f, 0.f};
  float c[4] = {0.f, 0.f, 0.f, 0.f};

  for (int s = 0; s < SLEN; ++s) {
    const float4 pv = *(const float4*)(pre + ((size_t)s * BATCH + b) * 16 + g * 4);
    float ang[4] = {pv.x, pv.y, pv.z, pv.w};
#pragma unroll
    for (int q = 0; q < 4; ++q)
      ang[q] += WhG[q * 4 + 0] * h[0] + WhG[q * 4 + 1] * h[1] +
                WhG[q * 4 + 2] * h[2] + WhG[q * 4 + 3] * h[3];

    float cw[4], sw[4];
#pragma unroll
    for (int q = 0; q < 4; ++q) __sincosf(0.5f * ang[q], &sw[q], &cw[q]);
    float p01[4] = {cw[0]*cw[1], cw[0]*sw[1], sw[0]*cw[1], sw[0]*sw[1]};
    float p23[4] = {cw[2]*cw[3], cw[2]*sw[3], sw[2]*cw[3], sw[2]*sw[3]};
    float s16[16];
#pragma unroll
    for (int i = 0; i < 16; ++i) s16[i] = p01[i >> 2] * p23[i & 3];

    float act[4];
#pragma unroll
    for (int w = 0; w < 4; ++w) {
      const float* Mw = Mg + w * 256;
      float z = 0.f;
#pragma unroll
      for (int jj = 0; jj < 16; ++jj) {
        float tj = 0.f;
#pragma unroll
        for (int k = 0; k < 16; ++k) tj += Mw[jj * 16 + k] * s16[k];
        z += s16[jj] * tj;
      }
      // gate 2 -> tanh(z) = 2*sigmoid(2z)-1 ; others -> sigmoid(z). One exp.
      float zz = (g == 2) ? 2.f * z : z;
      float r  = 1.f / (1.f + __expf(-zz));
      act[w] = (g == 2) ? (2.f * r - 1.f) : r;
    }

#pragma unroll
    for (int q = 0; q < 4; ++q) {
      float own = act[q];
      float v1 = __shfl_xor(own, 1, 32);
      float v2 = __shfl_xor(own, 2, 32);
      float v3 = __shfl_xor(v1, 2, 32);
      float fq = pick4(own, v1, v2, v3, g ^ 0);
      float iq = pick4(own, v1, v2, v3, g ^ 1);
      float gq = pick4(own, v1, v2, v3, g ^ 2);
      float oq = pick4(own, v1, v2, v3, g ^ 3);
      c[q] = fq * c[q] + iq * gq;
      float tc = 2.f / (1.f + __expf(-2.f * c[q])) - 1.f;
      h[q] = oq * tc;
    }
    if (g == 0)
      *(float4*)(outsH + ((size_t)s * BATCH + b) * 4) = make_float4(h[0], h[1], h[2], h[3]);
  }
}

// ---------------------------------------------------------------------------
// Kernel 4: logits + log_softmax. One row (s,b) per lane; tag weights in LDS.
// ---------------------------------------------------------------------------
__global__ void __launch_bounds__(256)
k_logits(const float* __restrict__ outsH, const float* __restrict__ tag_w,
         const float* __restrict__ tag_b, float* __restrict__ out)
{
  __shared__ float tw[TTAGS * 4];
  __shared__ float tb[TTAGS];
  if (threadIdx.x < TTAGS * 4) tw[threadIdx.x] = tag_w[threadIdx.x];
  if (threadIdx.x < TTAGS)     tb[threadIdx.x] = tag_b[threadIdx.x];
  __syncthreads();

  const int r = blockIdx.x * blockDim.x + threadIdx.x;   // 0..131071
  const float4 hv = *(const float4*)(outsH + (size_t)r * 4);
  float lg[TTAGS];
  float mx = -1e30f;
#pragma unroll
  for (int tt = 0; tt < TTAGS; ++tt) {
    float v = tb[tt] + tw[tt * 4 + 0] * hv.x + tw[tt * 4 + 1] * hv.y +
                       tw[tt * 4 + 2] * hv.z + tw[tt * 4 + 3] * hv.w;
    lg[tt] = v;
    mx = fmaxf(mx, v);
  }
  float sum = 0.f;
#pragma unroll
  for (int tt = 0; tt < TTAGS; ++tt) sum += __expf(lg[tt] - mx);
  const float lse = __logf(sum) + mx;
  float* orow = out + (size_t)r * TTAGS;
#pragma unroll
  for (int tt = 0; tt < TTAGS; tt += 4)
    *(float4*)(orow + tt) = make_float4(lg[tt] - lse, lg[tt + 1] - lse,
                                        lg[tt + 2] - lse, lg[tt + 3] - lse);
}

// ---------------------------------------------------------------------------
// Host: exact numpy MT19937 replication of RAND_OPS (deterministic, CPU-only,
// baked into kernargs -> graph-capture safe), then 4 kernel launches.
// ---------------------------------------------------------------------------
namespace {
struct HostMT {
  uint32_t mt[624]; int idx;
  explicit HostMT(uint32_t s) {
    mt[0] = s;
    for (int i = 1; i < 624; ++i)
      mt[i] = 1812433253u * (mt[i - 1] ^ (mt[i - 1] >> 30)) + (uint32_t)i;
    idx = 624;
  }
  uint32_t next32() {
    if (idx >= 624) {
      for (int i = 0; i < 624; ++i) {
        uint32_t y = (mt[i] & 0x80000000u) | (mt[(i + 1) % 624] & 0x7fffffffu);
        uint32_t v = mt[(i + 397) % 624] ^ (y >> 1);
        if (y & 1u) v ^= 2567483615u;
        mt[i] = v;
      }
      idx = 0;
    }
    uint32_t y = mt[idx++];
    y ^= y >> 11; y ^= (y << 7) & 2636928640u; y ^= (y << 15) & 4022730752u; y ^= y >> 18;
    return y;
  }
  uint64_t next64() { uint64_t hi = next32(); uint64_t lo = next32(); return (hi << 32) | lo; }
  double nextDouble() {
    uint32_t a = next32() >> 5, b = next32() >> 6;
    return (a * 67108864.0 + b) * (1.0 / 9007199254740992.0);
  }
  int64_t randint(int64_t n) {            // legacy _masked_uint64 path, [0, n)
    uint64_t rng = (uint64_t)n - 1;
    uint64_t mask = rng;
    mask |= mask >> 1; mask |= mask >> 2; mask |= mask >> 4;
    mask |= mask >> 8; mask |= mask >> 16; mask |= mask >> 32;
    uint64_t v;
    do { v = next64() & mask; } while (v > rng);
    return (int64_t)v;
  }
};
} // namespace

extern "C" void kernel_launch(void* const* d_in, const int* in_sizes, int n_in,
                              void* d_out, int out_size, void* d_ws, size_t ws_size,
                              hipStream_t stream) {
  (void)in_sizes; (void)n_in; (void)out_size; (void)ws_size;
  const int*   sentence = (const int*)  d_in[0];
  const float* emb      = (const float*)d_in[1];
  const float* lin_w    = (const float*)d_in[2];
  const float* lin_b    = (const float*)d_in[3];
  const float* qrx      = (const float*)d_in[4];
  const float* qry      = (const float*)d_in[5];
  const float* qrz      = (const float*)d_in[6];
  const float* qcrx     = (const float*)d_in[7];
  const float* tag_w    = (const float*)d_in[8];
  const float* tag_b    = (const float*)d_in[9];
  float* out = (float*)d_out;

  char* ws = (char*)d_ws;
  float* Mbuf  = (float*)ws;                                  // 16 KB
  float* pre   = (float*)(ws + 32768);                        // 8 MB
  float* outsH = (float*)(ws + 32768 + (size_t)8 * 1024 * 1024); // 2 MB

  // Replicate RAND_OPS = [_rand_ops(seed) for seed in 0..3]
  QOps ops;
  for (int seed = 0; seed < 4; ++seed) {
    HostMT rng((uint32_t)seed);
    for (int o = 0; o < 20; ++o) {
      int k = (int)rng.randint(4);
      if (k < 3) {
        int w = (int)rng.randint(NQ);
        double th = rng.nextDouble() * 6.283185307179586;
        ops.op[seed][o] = QOp{k, w, 0, (float)th};
      } else {
        int w0 = (int)rng.randint(NQ);
        int w1 = (int)((w0 + 1 + rng.randint(NQ - 1)) % NQ);
        ops.op[seed][o] = QOp{3, w0, w1, 0.f};
      }
    }
  }

  k_build_m<<<dim3(4),    dim3(32),  0, stream>>>(ops, qrx, qry, qrz, qcrx, Mbuf);
  k_pregemm<<<dim3(1024), dim3(256), 0, stream>>>(sentence, emb, lin_w, lin_b, pre);
  k_scan   <<<dim3(8),    dim3(256), 0, stream>>>(pre, lin_w, Mbuf, outsH);
  k_logits <<<dim3(512),  dim3(256), 0, stream>>>(outsH, tag_w, tag_b, out);
}
